// MbpGINEMessagePassing_24824910970957
// MI455X (gfx1250) — matrix operand order repaired and checked by reading, not compiled, not run on hardware
//
#include <hip/hip_runtime.h>
#include <hip/hip_bf16.h>

#define HIDDEN 128
#define ATTN   128

typedef __attribute__((ext_vector_type(16))) __bf16 bf16x16;
typedef __attribute__((ext_vector_type(8)))  __bf16 bf16x8;
typedef __attribute__((ext_vector_type(8)))  float  f32x8;
typedef __attribute__((ext_vector_type(4)))  float  f32x4;

// ---------------------------------------------------------------------------
// Convert the four 128x128 fp32 weight matrices to bf16 (one pass, tiny).
// ---------------------------------------------------------------------------
__global__ void gine_convert_weights(const float* __restrict__ Qw,
                                     const float* __restrict__ Kw,
                                     const float* __restrict__ Ew,
                                     const float* __restrict__ Cw,
                                     __bf16* __restrict__ qwb,
                                     __bf16* __restrict__ kwb,
                                     __bf16* __restrict__ ewb,
                                     __bf16* __restrict__ cwb) {
  int i = blockIdx.x * blockDim.x + threadIdx.x;
  if (i < ATTN * HIDDEN) {
    qwb[i] = (__bf16)Qw[i];
    kwb[i] = (__bf16)Kw[i];
    ewb[i] = (__bf16)Ew[i];
    cwb[i] = (__bf16)Cw[i];
  }
}

// ---------------------------------------------------------------------------
// Zero the agg region of d_out (harness poisons it; we scatter with atomics).
// ---------------------------------------------------------------------------
__global__ void gine_zero(float* __restrict__ p, long long n) {
  long long i = (long long)blockIdx.x * blockDim.x + threadIdx.x;
  long long stride = (long long)gridDim.x * blockDim.x;
  for (; i < n; i += stride) p[i] = 0.0f;
}

// Load a 16x32 bf16 A-operand chunk from a row-major fp32 matrix (ld = 128).
// A layout: lane holds row M = lane&15; half = lane>>4;
//   elements 0..7  -> K = k0 + half*8 + e
//   elements 8..15 -> K = k0 + 16 + half*8 + (e-8)
__device__ __forceinline__ bf16x16 load_a_f32(const float* __restrict__ rowp,
                                              int k0, int half) {
  const float* p = rowp + k0 + half * 8;
  f32x4 f0 = *(const f32x4*)(p + 0);
  f32x4 f1 = *(const f32x4*)(p + 4);
  f32x4 f2 = *(const f32x4*)(p + 16);
  f32x4 f3 = *(const f32x4*)(p + 20);
  bf16x16 a;
#pragma unroll
  for (int i = 0; i < 4; ++i) {
    a[i]      = (__bf16)f0[i];
    a[4 + i]  = (__bf16)f1[i];
    a[8 + i]  = (__bf16)f2[i];
    a[12 + i] = (__bf16)f3[i];
  }
  return a;
}

// Load a 32x16 bf16 B-operand chunk. B layout: lane holds column N = lane&15,
// elements 0..7 -> K = k0 + half*8 + e, elements 8..15 -> K = k0 + 16 + half*8 + (e-8).
// For row-major [out,in] weights these are two 16-byte loads from weight row N.
__device__ __forceinline__ bf16x16 load_b_bf16(const __bf16* __restrict__ w,
                                               int n, int k0, int half) {
  const __bf16* p = w + (size_t)n * HIDDEN + k0 + half * 8;
  bf16x8 lo = *(const bf16x8*)(p + 0);
  bf16x8 hi = *(const bf16x8*)(p + 16);
  bf16x16 b;
#pragma unroll
  for (int i = 0; i < 8; ++i) { b[i] = lo[i]; b[8 + i] = hi[i]; }
  return b;
}

// ---------------------------------------------------------------------------
// Node projections: Qx = x @ Qw^T + Qb, Kx = x @ Kw^T + Kb, fp32 out to ws.
// One wave per 16-node tile, 4 waves per block.
// ---------------------------------------------------------------------------
__global__ void gine_node_proj(const float* __restrict__ x,
                               const __bf16* __restrict__ qwb,
                               const float* __restrict__ Qb,
                               const __bf16* __restrict__ kwb,
                               const float* __restrict__ Kb,
                               float* __restrict__ Qx,
                               float* __restrict__ Kx,
                               int n_nodes) {
  const int wave = threadIdx.x >> 5;
  const int lane = threadIdx.x & 31;
  const int half = lane >> 4;
  const int row  = lane & 15;
  const int node0 = (blockIdx.x * 4 + wave) * 16;
  if (node0 >= n_nodes) return;  // wave-uniform

  const int nr = min(node0 + row, n_nodes - 1);
  const float* xrow = x + (size_t)nr * HIDDEN;

  bf16x16 a[4];
#pragma unroll
  for (int kc = 0; kc < 4; ++kc) a[kc] = load_a_f32(xrow, kc * 32, half);

#pragma unroll
  for (int nb = 0; nb < 8; ++nb) {
    const int colg = nb * 16 + row;  // output column (row var == lane&15)
    float qb = Qb[colg];
    float kb = Kb[colg];
    f32x8 cq, ck;
#pragma unroll
    for (int j = 0; j < 8; ++j) { cq[j] = qb; ck[j] = kb; }
#pragma unroll
    for (int kc = 0; kc < 4; ++kc) {
      bf16x16 bq = load_b_bf16(qwb, colg, kc * 32, half);
      cq = __builtin_amdgcn_wmma_f32_16x16x32_bf16(false, a[kc], false, bq,
                                                   (short)0, cq, false, false);
      bf16x16 bk = load_b_bf16(kwb, colg, kc * 32, half);
      ck = __builtin_amdgcn_wmma_f32_16x16x32_bf16(false, a[kc], false, bk,
                                                   (short)0, ck, false, false);
    }
    // C/D layout: VGPR j -> row j + half*8, col = lane&15 within this nb block
#pragma unroll
    for (int j = 0; j < 8; ++j) {
      int r = node0 + j + half * 8;
      if (r < n_nodes) {
        Qx[(size_t)r * ATTN + colg] = cq[j];
        Kx[(size_t)r * ATTN + colg] = ck[j];
      }
    }
  }
}

// ---------------------------------------------------------------------------
// Edge kernel: h = relu(Qx[dst] + Kx[src] + conn_feat @ Ew^T)   [16 x 128]
//              conn = h @ Cw^T + Cb                              [16 x 128]
//              conn -> d_out, atomically scattered into agg[dst].
// One wave per 16-edge tile, 4 waves per block.
// ---------------------------------------------------------------------------
#define HPAD 136  // 128 + 8 bf16 pad: shifts each row by 4 LDS banks

__global__ void gine_edge(const float* __restrict__ conn_feat,
                          const long long* __restrict__ edge_index,
                          const float* __restrict__ Qx,
                          const float* __restrict__ Kx,
                          const __bf16* __restrict__ ewb,
                          const __bf16* __restrict__ cwb,
                          const float* __restrict__ Cb,
                          float* __restrict__ agg,
                          float* __restrict__ conn,
                          int n_edges) {
  __shared__ int s_dst[4][16];
  __shared__ int s_src[4][16];
  __shared__ __attribute__((aligned(16))) __bf16 s_h[4][16][HPAD];

  const int wave = threadIdx.x >> 5;
  const int lane = threadIdx.x & 31;
  const int half = lane >> 4;
  const int row  = lane & 15;
  const int e0   = (blockIdx.x * 4 + wave) * 16;
  const bool active = (e0 < n_edges);  // wave-uniform

  if (active && lane < 16) {
    int ec = min(e0 + lane, n_edges - 1);
    s_dst[wave][lane] = (int)edge_index[ec];            // dst = edge_index[0]
    s_src[wave][lane] = (int)edge_index[n_edges + ec];  // src = edge_index[1]
  }
  __syncthreads();

  if (active) {
    // ---- cache A = conn_feat tile (16 x 128) as bf16 ----
    const int er = min(e0 + row, n_edges - 1);
    const float* frow = conn_feat + (size_t)er * HIDDEN;
    bf16x16 a[4];
#pragma unroll
    for (int kc = 0; kc < 4; ++kc) a[kc] = load_a_f32(frow, kc * 32, half);

    // ---- phase 1: h = relu(Qdst + Ksrc + A @ Ew^T), write bf16 to LDS ----
#pragma unroll
    for (int nb = 0; nb < 8; ++nb) {
      const int colg = nb * 16 + row;
      f32x8 c;
#pragma unroll
      for (int j = 0; j < 8; ++j) {
        int r = j + half * 8;
        int d = s_dst[wave][r];
        int s = s_src[wave][r];
        c[j] = Qx[(size_t)d * ATTN + colg] + Kx[(size_t)s * ATTN + colg];
      }
#pragma unroll
      for (int kc = 0; kc < 4; ++kc) {
        bf16x16 b = load_b_bf16(ewb, colg, kc * 32, half);
        c = __builtin_amdgcn_wmma_f32_16x16x32_bf16(false, a[kc], false, b,
                                                    (short)0, c, false, false);
      }
#pragma unroll
      for (int j = 0; j < 8; ++j) {
        float v = fmaxf(c[j], 0.0f);                    // ReLU
        s_h[wave][j + half * 8][colg] = (__bf16)v;
      }
    }
  }
  __syncthreads();

  if (active) {
    // ---- re-load h from LDS in A-operand layout (ds_load_b128 x2/chunk) ----
    bf16x16 a2[4];
#pragma unroll
    for (int kc = 0; kc < 4; ++kc) {
      const __bf16* p = &s_h[wave][row][kc * 32 + half * 8];
      bf16x8 lo = *(const bf16x8*)(p + 0);
      bf16x8 hi = *(const bf16x8*)(p + 16);
#pragma unroll
      for (int i = 0; i < 8; ++i) { a2[kc][i] = lo[i]; a2[kc][8 + i] = hi[i]; }
    }

    // ---- phase 2: conn = h @ Cw^T + Cb; store conn + atomic scatter ----
#pragma unroll
    for (int nb = 0; nb < 8; ++nb) {
      const int colg = nb * 16 + row;
      float cb = Cb[colg];
      f32x8 c;
#pragma unroll
      for (int j = 0; j < 8; ++j) c[j] = cb;
#pragma unroll
      for (int kc = 0; kc < 4; ++kc) {
        bf16x16 b = load_b_bf16(cwb, colg, kc * 32, half);
        c = __builtin_amdgcn_wmma_f32_16x16x32_bf16(false, a2[kc], false, b,
                                                    (short)0, c, false, false);
      }
#pragma unroll
      for (int j = 0; j < 8; ++j) {
        int r = j + half * 8;
        int e = e0 + r;
        if (e < n_edges) {
          conn[(size_t)e * HIDDEN + colg] = c[j];
          atomicAdd(&agg[(size_t)s_dst[wave][r] * HIDDEN + colg], c[j]);
        }
      }
    }
  }
}

// ---------------------------------------------------------------------------
// Launch
// ---------------------------------------------------------------------------
extern "C" void kernel_launch(void* const* d_in, const int* in_sizes, int n_in,
                              void* d_out, int out_size, void* d_ws, size_t ws_size,
                              hipStream_t stream) {
  const float*     x         = (const float*)d_in[0];
  const float*     conn_feat = (const float*)d_in[1];
  const long long* edge_idx  = (const long long*)d_in[2];  // int64 [2, E]
  const float*     Qw        = (const float*)d_in[3];
  const float*     Qb        = (const float*)d_in[4];
  const float*     Kw        = (const float*)d_in[5];
  const float*     Kb        = (const float*)d_in[6];
  const float*     Ew        = (const float*)d_in[7];
  const float*     Cw        = (const float*)d_in[8];
  const float*     Cb        = (const float*)d_in[9];

  const int n_nodes = in_sizes[0] / HIDDEN;
  const int n_edges = in_sizes[1] / HIDDEN;

  float* agg  = (float*)d_out;                       // [N, 128]
  float* conn = agg + (size_t)n_nodes * HIDDEN;      // [E, 128]

  // Workspace layout (all offsets 256B aligned):
  //   Qx fp32 [N,128] | Kx fp32 [N,128] | Qw/Kw/Ew/Cw bf16 [128,128] each
  char*   ws  = (char*)d_ws;
  float*  Qx  = (float*)ws;
  float*  Kx  = (float*)(ws + (size_t)n_nodes * HIDDEN * 4);
  __bf16* qwb = (__bf16*)(ws + (size_t)n_nodes * HIDDEN * 8);
  __bf16* kwb = qwb + ATTN * HIDDEN;
  __bf16* ewb = kwb + ATTN * HIDDEN;
  __bf16* cwb = ewb + ATTN * HIDDEN;

  gine_convert_weights<<<(ATTN * HIDDEN + 255) / 256, 256, 0, stream>>>(
      Qw, Kw, Ew, Cw, qwb, kwb, ewb, cwb);

  gine_zero<<<2048, 256, 0, stream>>>(agg, (long long)n_nodes * HIDDEN);

  const int ntiles = (n_nodes + 15) / 16;
  gine_node_proj<<<(ntiles + 3) / 4, 128, 0, stream>>>(
      x, qwb, Qb, kwb, Kb, Qx, Kx, n_nodes);

  const int etiles = (n_edges + 15) / 16;
  gine_edge<<<(etiles + 3) / 4, 128, 0, stream>>>(
      conn_feat, edge_idx, Qx, Kx, ewb, cwb, Cb, agg, conn, n_edges);
}
